// AttentionHead_52037823758572
// MI455X (gfx1250) — matrix operand order repaired and checked by reading, not compiled, not run on hardware
//
#include <hip/hip_runtime.h>
#include <hip/hip_fp16.h>

typedef __attribute__((ext_vector_type(16))) _Float16 v16h;
typedef __attribute__((ext_vector_type(8)))  float    v8f;

#define WPB 2            // waves per block in the tile kernels
#define EPS_LN 1e-5f

// ---------------------------------------------------------------------------
// WMMA helper:  D = A(16x32 f16) * B(32x16 f16) + C(16x16 f32)
// ---------------------------------------------------------------------------
__device__ __forceinline__ v8f wmma_f16(v16h a, v16h b, v8f c) {
  return __builtin_amdgcn_wmma_f32_16x16x32_f16(
      /*neg_a=*/false, a, /*neg_b=*/false, b,
      /*c_mod=*/(short)0, c, /*reuse_a=*/false, /*reuse_b=*/false);
}

__device__ __forceinline__ float wave_sum32(float v) {
  v += __shfl_xor(v, 16, 32);
  v += __shfl_xor(v,  8, 32);
  v += __shfl_xor(v,  4, 32);
  v += __shfl_xor(v,  2, 32);
  v += __shfl_xor(v,  1, 32);
  return v;
}

// Assemble one 16x32 f16 A-fragment directly from an f32 row in memory.
// Per ISA 7.12.2: lane (half = lane>>4) holds K = half*8 + e (e<8) and
// K = 16 + half*8 + (e-8) (e>=8)  -> two contiguous runs of 8 floats at
// p[0..7] and p[16..23], where p = row_base + kt*32 + half*8.
__device__ __forceinline__ v16h make_a_frag(const float* __restrict__ p) {
  float4 f0 = ((const float4*)p)[0];
  float4 f1 = ((const float4*)p)[1];
  float4 f2 = ((const float4*)(p + 16))[0];
  float4 f3 = ((const float4*)(p + 16))[1];
  v16h a;
  a[0]  = (_Float16)f0.x; a[1]  = (_Float16)f0.y; a[2]  = (_Float16)f0.z; a[3]  = (_Float16)f0.w;
  a[4]  = (_Float16)f1.x; a[5]  = (_Float16)f1.y; a[6]  = (_Float16)f1.z; a[7]  = (_Float16)f1.w;
  a[8]  = (_Float16)f2.x; a[9]  = (_Float16)f2.y; a[10] = (_Float16)f2.z; a[11] = (_Float16)f2.w;
  a[12] = (_Float16)f3.x; a[13] = (_Float16)f3.y; a[14] = (_Float16)f3.z; a[15] = (_Float16)f3.w;
  return a;
}

// ---------------------------------------------------------------------------
// Kernel 0: pack an f32 [K x N] weight matrix into f16 B-fragment order.
// B layout (per sparse-B table, restricted to K=32): lane&15 = N within tile,
// K-local = (lane>>4)*8 + (e&7) + (e>>3)*16.  Fragment (kt,nt) is 512 contiguous f16.
// ---------------------------------------------------------------------------
__global__ void pack_b_kernel(const float* __restrict__ src, _Float16* __restrict__ dst,
                              int K, int N, int ntiles) {
  int o = blockIdx.x * blockDim.x + threadIdx.x;
  if (o >= K * N) return;
  int e    = o & 15;
  int lane = (o >> 4) & 31;
  int frag = o >> 9;
  int nt   = frag % ntiles;
  int kt   = frag / ntiles;
  int n    = nt * 16 + (lane & 15);
  int k    = kt * 32 + (lane >> 4) * 8 + (e & 7) + ((e >> 3) << 4);
  dst[o] = (_Float16)src[k * N + n];
}

// ---------------------------------------------------------------------------
// Kernel 1: Q,K,V = node_fea[N,128] @ W[128,64] + b    (WMMA, one wave/16 rows)
// ---------------------------------------------------------------------------
__global__ void qkv_kernel(const float* __restrict__ X,
                           const _Float16* __restrict__ WQp, const _Float16* __restrict__ WKp,
                           const _Float16* __restrict__ WVp,
                           const float* __restrict__ bQ, const float* __restrict__ bK,
                           const float* __restrict__ bV,
                           float* __restrict__ Qw, float* __restrict__ Kw, float* __restrict__ Vw,
                           int nTiles) {
  int wid = (blockIdx.x * blockDim.x + threadIdx.x) >> 5;
  if (wid >= nTiles) return;
  int lane = threadIdx.x & 31, l15 = lane & 15, half = lane >> 4;
  int arow = wid * 16 + l15;           // A-fragment row for this lane

  v8f aQ[4], aK[4], aV[4];
#pragma unroll
  for (int nt = 0; nt < 4; nt++) { aQ[nt] = (v8f){0,0,0,0,0,0,0,0}; aK[nt] = aQ[nt]; aV[nt] = aQ[nt]; }

#pragma unroll
  for (int kt = 0; kt < 4; kt++) {
    v16h a = make_a_frag(X + arow * 128 + kt * 32 + half * 8);
#pragma unroll
    for (int nt = 0; nt < 4; nt++) {
      int off = ((kt * 4 + nt) * 32 + lane) * 16;
      aQ[nt] = wmma_f16(a, *(const v16h*)(WQp + off), aQ[nt]);
      aK[nt] = wmma_f16(a, *(const v16h*)(WKp + off), aK[nt]);
      aV[nt] = wmma_f16(a, *(const v16h*)(WVp + off), aV[nt]);
    }
  }

#pragma unroll
  for (int nt = 0; nt < 4; nt++) {
    int col = nt * 16 + l15;
    float bq = bQ[col], bk = bK[col], bv = bV[col];
#pragma unroll
    for (int r = 0; r < 8; r++) {
      int row = wid * 16 + r + half * 8;
      Qw[row * 64 + col] = aQ[nt][r] + bq;
      Kw[row * 64 + col] = aK[nt][r] + bk;
      Vw[row * 64 + col] = aV[nt][r] + bv;
    }
  }
}

// ---------------------------------------------------------------------------
// Kernel 2: fused per-edge pipeline, one wave per 16-edge tile.
// ---------------------------------------------------------------------------
struct WaveSmem {
  float Ep[16 * 66];                    // Ep tile, stride 66
  float GP[16 * 196];                   // gate[16][192] then p[16][132] (reuse)
  __align__(32) _Float16 FA[6 * 32 * 16]; // f16 A-fragments (zij, then mij)
};

__global__ void __launch_bounds__(WPB * 32)
edge_kernel(const int* __restrict__ idx1, const int* __restrict__ idx2,
            const float* __restrict__ edge_fea,
            const _Float16* __restrict__ WEp, const float* __restrict__ bE,
            const _Float16* __restrict__ WuP, const float* __restrict__ bu,
            const _Float16* __restrict__ WmP, const float* __restrict__ bm,
            const float* __restrict__ g1, const float* __restrict__ b1,
            const float* __restrict__ g2, const float* __restrict__ b2,
            const float* __restrict__ Qw, const float* __restrict__ Kw,
            const float* __restrict__ Vw,
            float* __restrict__ out, int nTiles) {
  __shared__ WaveSmem smem[WPB];
  int wslot = threadIdx.x >> 5;
  int wid   = blockIdx.x * WPB + wslot;
  if (wid >= nTiles) return;

  int lane = threadIdx.x & 31, l15 = lane & 15, half = lane >> 4;
  int base = wid * 16;
  float*    Ep = smem[wslot].Ep;
  float*    GP = smem[wslot].GP;
  _Float16* FA = smem[wslot].FA;

  // ---- Ep = edge_fea_tile[16x64] @ WE[64x64] + bE  (WMMA) --------------
  {
    v8f acc[4];
#pragma unroll
    for (int nt = 0; nt < 4; nt++) acc[nt] = (v8f){0,0,0,0,0,0,0,0};
    int row = base + l15;
#pragma unroll
    for (int kt = 0; kt < 2; kt++) {
      v16h a = make_a_frag(edge_fea + row * 64 + kt * 32 + half * 8);
#pragma unroll
      for (int nt = 0; nt < 4; nt++)
        acc[nt] = wmma_f16(a, *(const v16h*)(WEp + ((kt * 4 + nt) * 32 + lane) * 16), acc[nt]);
    }
#pragma unroll
    for (int nt = 0; nt < 4; nt++) {
      int col = nt * 16 + l15;
      float bb = bE[col];
#pragma unroll
      for (int r = 0; r < 8; r++) Ep[(r + half * 8) * 66 + col] = acc[nt][r] + bb;
    }
  }

  // indices: lanes 0-15 carry idx1, lanes 16-31 carry idx2 of the tile
  int iv = (lane < 16) ? idx1[base + l15] : idx2[base + l15];

  const float invs = 0.07216878364870322f;  // 1/sqrt(192)

  // ---- hoist per-lane loop invariants for phase A -----------------------
  // Lane handles columns c = lane + 32*t (t=0..5). Precompute LN params and
  // fragment-scatter addresses; only the node gathers stay in the j loop.
  float g1v[6], b1v[6];
  int   segv[6], offv[6], fragBase[6];
#pragma unroll
  for (int t = 0; t < 6; t++) {
    int c = lane + t * 32;
    g1v[t] = g1[c];
    b1v[t] = b1[c];
    segv[t] = c >> 6;
    offv[t] = c & 63;
    int kt = c >> 5, kl = c & 31;
    int hh = (kl >> 3) & 1;
    int ee = (kl & 7) + ((kl >> 4) << 3);
    fragBase[t] = (kt * 32 + hh * 16) * 16 + ee;   // + j*16 per edge
  }

  // ---- phase A: aij -> LN -> sigmoid gate; zij -> f16 A-fragments -------
  for (int j = 0; j < 16; j++) {
    int i1 = __shfl(iv, j, 32);
    int i2 = __shfl(iv, j + 16, 32);
    float av[6], zv[6];
    float s = 0.f, s2 = 0.f;
#pragma unroll
    for (int t = 0; t < 6; t++) {
      int off = offv[t];
      float q = Qw[i1 * 64 + off];
      float kk, zz;
      if (segv[t] == 0)      { kk = Kw[i1 * 64 + off]; zz = Vw[i1 * 64 + off]; }
      else if (segv[t] == 1) { kk = Kw[i2 * 64 + off]; zz = Vw[i2 * 64 + off]; }
      else                   { float e = Ep[j * 66 + off]; kk = e; zz = e; }
      float a = q * kk * invs;
      av[t] = a; zv[t] = zz;
      s += a; s2 += a * a;
    }
    s  = wave_sum32(s);
    s2 = wave_sum32(s2);
    float mean = s * (1.0f / 192.0f);
    float var  = s2 * (1.0f / 192.0f) - mean * mean;
    float rstd = rsqrtf(var + EPS_LN);
#pragma unroll
    for (int t = 0; t < 6; t++) {
      float ln   = (av[t] - mean) * rstd * g1v[t] + b1v[t];
      float gate = 1.0f / (1.0f + __expf(-ln));
      GP[j * 196 + lane + t * 32] = gate;
      FA[fragBase[t] + j * 16] = (_Float16)zv[t];   // zij scatter, A-frag order
    }
  }

  // ---- GEMM1: u = zij[16x192] @ Wu[192x192]  (72 WMMA) ------------------
  v8f u[12];
#pragma unroll
  for (int nt = 0; nt < 12; nt++) u[nt] = (v8f){0,0,0,0,0,0,0,0};
#pragma unroll
  for (int kt = 0; kt < 6; kt++) {
    v16h a = *(const v16h*)(FA + (kt * 32 + lane) * 16);
#pragma unroll
    for (int nt = 0; nt < 12; nt++)
      u[nt] = wmma_f16(a, *(const v16h*)(WuP + ((kt * 12 + nt) * 32 + lane) * 16), u[nt]);
  }

  // ---- mij = gate * (u + bu) -> repack as A-fragments (reuse FA) --------
#pragma unroll
  for (int nt = 0; nt < 12; nt++) {
    int col = nt * 16 + l15;
    float bb = bu[col];
    int kt = col >> 5, kl = col & 31;
    int hh = (kl >> 3) & 1;
    int ee = (kl & 7) + ((kl >> 4) << 3);
#pragma unroll
    for (int r = 0; r < 8; r++) {
      int row = r + half * 8;
      float m = GP[row * 196 + col] * (u[nt][r] + bb);
      FA[(kt * 32 + hh * 16 + row) * 16 + ee] = (_Float16)m;
    }
  }

  // ---- GEMM2: p = mij[16x192] @ Wm[192x128]  (48 WMMA) ------------------
  v8f p[8];
#pragma unroll
  for (int nt = 0; nt < 8; nt++) p[nt] = (v8f){0,0,0,0,0,0,0,0};
#pragma unroll
  for (int kt = 0; kt < 6; kt++) {
    v16h a = *(const v16h*)(FA + (kt * 32 + lane) * 16);
#pragma unroll
    for (int nt = 0; nt < 8; nt++)
      p[nt] = wmma_f16(a, *(const v16h*)(WmP + ((kt * 8 + nt) * 32 + lane) * 16), p[nt]);
  }
#pragma unroll
  for (int nt = 0; nt < 8; nt++) {
    int col = nt * 16 + l15;
    float bb = bm[col];
#pragma unroll
    for (int r = 0; r < 8; r++) GP[(r + half * 8) * 132 + col] = p[nt][r] + bb;
  }

  // ---- phase C: LN over 128, atomic segment-sum scatter -----------------
  float g2v[4], b2v[4];
#pragma unroll
  for (int t = 0; t < 4; t++) { g2v[t] = g2[lane + t * 32]; b2v[t] = b2[lane + t * 32]; }

  for (int j = 0; j < 16; j++) {
    int i1 = __shfl(iv, j, 32);
    float v0[4];
    float s = 0.f, s2 = 0.f;
#pragma unroll
    for (int t = 0; t < 4; t++) {
      float v = GP[j * 132 + lane + t * 32];
      v0[t] = v; s += v; s2 += v * v;
    }
    s  = wave_sum32(s);
    s2 = wave_sum32(s2);
    float mean = s * (1.0f / 128.0f);
    float var  = s2 * (1.0f / 128.0f) - mean * mean;
    float rstd = rsqrtf(var + EPS_LN);
#pragma unroll
    for (int t = 0; t < 4; t++) {
      float msg = (v0[t] - mean) * rstd * g2v[t] + b2v[t];
      unsafeAtomicAdd(&out[i1 * 128 + lane + t * 32], msg);  // global_atomic_add_f32
    }
  }
}

// ---------------------------------------------------------------------------
extern "C" void kernel_launch(void* const* d_in, const int* in_sizes, int n_in,
                              void* d_out, int out_size, void* d_ws, size_t ws_size,
                              hipStream_t stream) {
  (void)n_in; (void)ws_size;
  const float* node_fea = (const float*)d_in[0];
  const int*   idx1     = (const int*)d_in[1];
  const int*   idx2     = (const int*)d_in[2];
  const float* edge_fea = (const float*)d_in[3];
  const float* WQ = (const float*)d_in[4];  const float* bQ = (const float*)d_in[5];
  const float* WK = (const float*)d_in[6];  const float* bK = (const float*)d_in[7];
  const float* WV = (const float*)d_in[8];  const float* bV = (const float*)d_in[9];
  const float* WE = (const float*)d_in[10]; const float* bE = (const float*)d_in[11];
  const float* Wu = (const float*)d_in[12]; const float* bu = (const float*)d_in[13];
  const float* Wm = (const float*)d_in[14]; const float* bm = (const float*)d_in[15];
  const float* g1 = (const float*)d_in[16]; const float* b1 = (const float*)d_in[17];
  const float* g2 = (const float*)d_in[18]; const float* b2 = (const float*)d_in[19];

  const int N = in_sizes[0] / 128;   // 50000
  const int E = in_sizes[1];         // 800000

  // workspace layout
  float*     Qw  = (float*)d_ws;
  float*     Kw  = Qw + (size_t)N * 64;
  float*     Vw  = Kw + (size_t)N * 64;
  _Float16*  WQp = (_Float16*)(Vw + (size_t)N * 64);
  _Float16*  WKp = WQp + 128 * 64;
  _Float16*  WVp = WKp + 128 * 64;
  _Float16*  WEp = WVp + 128 * 64;
  _Float16*  WuP = WEp + 64 * 64;
  _Float16*  WmP = WuP + 192 * 192;

  hipMemsetAsync(d_out, 0, (size_t)out_size * sizeof(float), stream);

  pack_b_kernel<<<(128 * 64 + 255) / 256, 256, 0, stream>>>(WQ, WQp, 128, 64, 4);
  pack_b_kernel<<<(128 * 64 + 255) / 256, 256, 0, stream>>>(WK, WKp, 128, 64, 4);
  pack_b_kernel<<<(128 * 64 + 255) / 256, 256, 0, stream>>>(WV, WVp, 128, 64, 4);
  pack_b_kernel<<<(64  * 64 + 255) / 256, 256, 0, stream>>>(WE, WEp, 64, 64, 4);
  pack_b_kernel<<<(192 * 192 + 255) / 256, 256, 0, stream>>>(Wu, WuP, 192, 192, 12);
  pack_b_kernel<<<(192 * 128 + 255) / 256, 256, 0, stream>>>(Wm, WmP, 192, 128, 8);

  int nTilesN = (N + 15) / 16;
  qkv_kernel<<<(nTilesN + WPB - 1) / WPB, WPB * 32, 0, stream>>>(
      node_fea, WQp, WKp, WVp, bQ, bK, bV, Qw, Kw, Vw, nTilesN);

  int nTilesE = (E + 15) / 16;
  edge_kernel<<<(nTilesE + WPB - 1) / WPB, WPB * 32, 0, stream>>>(
      idx1, idx2, edge_fea, WEp, bE, WuP, bu, WmP, bm,
      g1, b1, g2, b2, Qw, Kw, Vw, (float*)d_out, nTilesE);
}
